// transformer_head_55791625175535
// MI455X (gfx1250) — matrix operand order repaired and checked by reading, not compiled, not run on hardware
//
#include <hip/hip_runtime.h>
#include <hip/hip_bf16.h>

// Point-transformer vector-attention head for MI455X (gfx1250, wave32).
// B=8 N=8192 K=16 C=64 POS_HID=64 F=4C=256.
// One wave == one point's 16 neighbors (M=16 WMMA tile); each wave loops over
// PPW points so the LDS weight staging is amortized 64 points per workgroup.
// Transposed GEMMs: weights = A operand (feature-major LDS, contiguous b128),
// activations = B operand (M-tile invariant -> register-hoisted), weight A
// tiles double-buffered, BN affine folded to precomputed scale/offset.

typedef __attribute__((ext_vector_type(16))) _Float16 v16h;
typedef __attribute__((ext_vector_type(8)))  _Float16 v8h;
typedef __attribute__((ext_vector_type(4)))  _Float16 v4h;
typedef __attribute__((ext_vector_type(8)))  float    v8f;

union V16H { v16h v; v8h q[2]; };

#define BB 8
#define NN 8192
#define KKN 16
#define CC 64
#define FF 256
#define WAVES 4
#define PPW 16            /* points per wave: 64 points per workgroup */
#define INVS 0.99950037f  /* 1/sqrt(1 + 1e-3)  (BN with init moving stats) */

// A operand (16x32 f16): lane (m = lane&15, hl = lane>>4) holds halfs
// [kOff+hl*8 .. +7] and [kOff+16+hl*8 .. +7] -> two contiguous 16B chunks.
// Here A = a 16-row tile of a feature-major weight matrix.
__device__ __forceinline__ v16h load_a16(const _Float16* base, int Kd, int lane, int kOff) {
  const int m = lane & 15, hl = lane >> 4;
  const _Float16* r = base + m * Kd + kOff + hl * 8;   // 16B aligned
  V16H u;
  u.q[0] = *(const v8h*)(r);
  u.q[1] = *(const v8h*)(r + 16);
  return u.v;
}

// B operand (32x16 f16): lane (n = lane&15, hl = lane>>4) holds halfs
// [kOff+hl*8 .. +7] and [kOff+16+hl*8 .. +7] -> two contiguous 16B chunks.
// Here B = activations stored row-major act[row][k] with row = n.
__device__ __forceinline__ v16h load_b16(const _Float16* act, int Kd, int lane, int kOff) {
  const int nn = lane & 15, hl = lane >> 4;
  const _Float16* r = act + nn * Kd + kOff + hl * 8;   // 16B aligned
  V16H u;
  u.q[0] = *(const v8h*)(r);
  u.q[1] = *(const v8h*)(r + 16);
  return u.v;
}

__device__ __forceinline__ v8f wmma16(v16h a, v16h b, v8f c) {
  return __builtin_amdgcn_wmma_f32_16x16x32_f16(false, a, false, b, (short)0, c, false, false);
}

extern "C" __global__ __launch_bounds__(128, 1)
void pt_attn_wmma(const float* __restrict__ key, const float* __restrict__ query,
                  const float* __restrict__ value, const float* __restrict__ pos,
                  const int* __restrict__ idx,
                  const float* __restrict__ W1, const float* __restrict__ b1,
                  const float* __restrict__ g1, const float* __restrict__ bt1,
                  const float* __restrict__ W2, const float* __restrict__ b2,
                  const float* __restrict__ W3, const float* __restrict__ b3,
                  const float* __restrict__ g3, const float* __restrict__ bt3,
                  const float* __restrict__ W4, const float* __restrict__ b4,
                  float* __restrict__ out)
{
  extern __shared__ char smem[];
  // ---- weight region (shared by all waves), f16, feature-major ----
  _Float16* W2t = (_Float16*)smem;            // [64][64]   8192 B
  _Float16* W3t = (_Float16*)(smem + 8192);   // [256][64]  32768 B
  _Float16* W4t = (_Float16*)(smem + 40960);  // [64][256]  32768 B
  float* sPar = (float*)(smem + 73728);       // packed f32 params, 960 floats
  float* pW1  = sPar;          // 192 (W1[d][c], d-major)
  float* pS1  = sPar + 192;    // 64  g1*invs
  float* pO1  = sPar + 256;    // 64  g1*b1*invs + bt1
  float* pb2  = sPar + 320;    // 64
  float* pb4s = sPar + 384;    // 64  b4*0.125
  float* pS3  = sPar + 448;    // 256 g3*invs
  float* pO3  = sPar + 704;    // 256 g3*b3*invs + bt3   -> ends at 960 floats

  const int tid  = threadIdx.x;
  const int w    = tid >> 5;
  const int lane = tid & 31;

  // ---- per-wave staging (20480 B each), base 77824 ----
  char* wb = smem + 77824 + w * 20480;
  float*    sQK  = (float*)wb;               // [16][64] f32  (reused: logits -> softmax a)
  float*    sVal = (float*)(wb + 4096);      // [16][64] f32  (in-place becomes v + pos_em)
  _Float16* sHA  = (_Float16*)(wb + 8192);   // [16][64] f16  pos-MLP hidden
  _Float16* sA3  = (_Float16*)(wb + 10240);  // [16][64] f16  qk_rel + pos_em
  _Float16* sAF  = (_Float16*)(wb + 12288);  // [16][256] f16 attn hidden

  // ================= Phase 0: weights -> LDS (f16, feature-major), once per WG =====
  for (int t = tid; t < 64 * 16; t += 128) {            // W2t[n][k] = W2[k][n]
    const int n = t >> 4, k = (t & 15) * 4;
    v4h h; h.x = (_Float16)W2[(k + 0) * 64 + n]; h.y = (_Float16)W2[(k + 1) * 64 + n];
           h.z = (_Float16)W2[(k + 2) * 64 + n]; h.w = (_Float16)W2[(k + 3) * 64 + n];
    *(v4h*)(W2t + n * 64 + k) = h;
  }
  for (int t = tid; t < 256 * 16; t += 128) {           // W3t[f][k] = W3[k][f]
    const int f = t >> 4, k = (t & 15) * 4;
    v4h h; h.x = (_Float16)W3[(k + 0) * 256 + f]; h.y = (_Float16)W3[(k + 1) * 256 + f];
           h.z = (_Float16)W3[(k + 2) * 256 + f]; h.w = (_Float16)W3[(k + 3) * 256 + f];
    *(v4h*)(W3t + f * 64 + k) = h;
  }
  for (int t = tid; t < 64 * 64; t += 128) {            // W4t[c][f] = W4[f][c]
    const int c = t >> 6, f = (t & 63) * 4;
    v4h h; h.x = (_Float16)W4[(f + 0) * 64 + c]; h.y = (_Float16)W4[(f + 1) * 64 + c];
           h.z = (_Float16)W4[(f + 2) * 64 + c]; h.w = (_Float16)W4[(f + 3) * 64 + c];
    *(v4h*)(W4t + c * 256 + f) = h;
  }
  for (int t = tid; t < 192; t += 128) pW1[t] = W1[t];
  if (tid < 64) {
    const float g = g1[tid];
    pS1[tid]  = g * INVS;
    pO1[tid]  = g * b1[tid] * INVS + bt1[tid];
    pb2[tid]  = b2[tid];
    pb4s[tid] = b4[tid] * 0.125f;          // 1/sqrt(C) folded
  }
  for (int t = tid; t < 256; t += 128) {
    const float g = g3[t];
    pS3[t] = g * INVS;
    pO3[t] = g * b3[t] * INVS + bt3[t];
  }
  __syncthreads();   // only cross-wave sync; everything below is wave-local

  const int r  = lane >> 1, hf = lane & 1;   // phase-1/5 map: row = neighbor k
  const int row = lane & 15;                 // D-tile: lane = point row
  const int hl  = lane >> 4;                 // D-tile: VGPR block = feature half

  // rows are (b,n,k) with k fastest, K==16 -> one wave tile == one point.
  // Each wave processes PPW consecutive points.
#pragma unroll 1
  for (int pp = 0; pp < PPW; ++pp) {
    const int p = (blockIdx.x * WAVES + w) * PPW + pp;   // p = b*N + n
    const size_t pOff = (size_t)p * CC;
    const int bIdx = p >> 13;                            // p / N

    // ================= Phase 1: gather + pos-MLP layer 1 =================
    const int nb = idx[(size_t)p * KKN + r];
    const size_t nbOff = ((size_t)bIdx * NN + nb) * CC;
    const size_t posP = (size_t)p * 3;
    const size_t posJ = ((size_t)bIdx * NN + nb) * 3;
    const float pr0 = pos[posP + 0] - pos[posJ + 0];
    const float pr1 = pos[posP + 1] - pos[posJ + 1];
    const float pr2 = pos[posP + 2] - pos[posJ + 2];

    const float4* q4 = (const float4*)(query + pOff) + hf * 8;
    const float4* k4 = (const float4*)(key + nbOff) + hf * 8;
    const float4* v4 = (const float4*)(value + pOff) + hf * 8;
    float4* qkOut = (float4*)(sQK + r * 64) + hf * 8;
    float4* vOut  = (float4*)(sVal + r * 64) + hf * 8;
#pragma unroll
    for (int i = 0; i < 8; ++i) {
      float4 q = q4[i], kk = k4[i], vv = v4[i];
      qkOut[i] = make_float4(q.x - kk.x, q.y - kk.y, q.z - kk.z, q.w - kk.w);
      vOut[i]  = vv;
    }
#pragma unroll
    for (int g = 0; g < 4; ++g) {                  // 8 channels per packed store
      v8h hh;
#pragma unroll
      for (int i = 0; i < 8; ++i) {
        const int c = hf * 32 + g * 8 + i;
        const float d = pr0 * pW1[c] + pr1 * pW1[64 + c] + pr2 * pW1[128 + c];
        hh[i] = (_Float16)fmaxf(pS1[c] * d + pO1[c], 0.f);
      }
      *(v8h*)(sHA + r * 64 + hf * 32 + g * 8) = hh;
    }
    // All LDS producer->consumer below is same-wave; DS ops are in-order per wave.

    // ================= Phase 2: pos_em^T = W2-tile @ h^T (M=channel, N=row) ========
    {
      const v16h b0 = load_b16(sHA, 64, lane, 0);   // activation B: nt-invariant
      const v16h b1v = load_b16(sHA, 64, lane, 32);
      v16h a0n = load_a16(W2t, 64, lane, 0);        // weight A: double-buffered
      v16h a1n = load_a16(W2t, 64, lane, 32);
#pragma unroll
      for (int nt = 0; nt < 4; ++nt) {
        const v16h a0 = a0n, a1 = a1n;
        if (nt < 3) {
          const _Float16* nxt = W2t + (nt + 1) * 16 * 64;
          a0n = load_a16(nxt, 64, lane, 0);
          a1n = load_a16(nxt, 64, lane, 32);
        }
        v8f acc = {};
        acc = wmma16(a0, b0, acc);
        acc = wmma16(a1, b1v, acc);
        const int cb = nt * 16 + hl * 8;            // 8 contiguous channels
        const float4 bb0 = *(const float4*)(pb2 + cb);
        const float4 bb1 = *(const float4*)(pb2 + cb + 4);
        const float pe[8] = { acc[0] + bb0.x, acc[1] + bb0.y, acc[2] + bb0.z, acc[3] + bb0.w,
                              acc[4] + bb1.x, acc[5] + bb1.y, acc[6] + bb1.z, acc[7] + bb1.w };
        const float4 q0 = *(const float4*)(sQK + row * 64 + cb);
        const float4 q1 = *(const float4*)(sQK + row * 64 + cb + 4);
        v8h hh;
        hh[0] = (_Float16)(q0.x + pe[0]); hh[1] = (_Float16)(q0.y + pe[1]);
        hh[2] = (_Float16)(q0.z + pe[2]); hh[3] = (_Float16)(q0.w + pe[3]);
        hh[4] = (_Float16)(q1.x + pe[4]); hh[5] = (_Float16)(q1.y + pe[5]);
        hh[6] = (_Float16)(q1.z + pe[6]); hh[7] = (_Float16)(q1.w + pe[7]);
        *(v8h*)(sA3 + row * 64 + cb) = hh;          // qk_rel + pos_em (f16)
        float4 v0 = *(float4*)(sVal + row * 64 + cb);
        float4 v1 = *(float4*)(sVal + row * 64 + cb + 4);
        v0.x += pe[0]; v0.y += pe[1]; v0.z += pe[2]; v0.w += pe[3];
        v1.x += pe[4]; v1.y += pe[5]; v1.z += pe[6]; v1.w += pe[7];
        *(float4*)(sVal + row * 64 + cb) = v0;      // v + pos_em (f32)
        *(float4*)(sVal + row * 64 + cb + 4) = v1;
      }
    }

    // ================= Phase 3: a^T = relu(BN(W3-tile @ (qk+pe)^T)) ================
    {
      const v16h b0 = load_b16(sA3, 64, lane, 0);   // activation B: nt-invariant
      const v16h b1v = load_b16(sA3, 64, lane, 32);
      v16h a0n = load_a16(W3t, 64, lane, 0);
      v16h a1n = load_a16(W3t, 64, lane, 32);
#pragma unroll
      for (int nt = 0; nt < 16; ++nt) {
        const v16h a0 = a0n, a1 = a1n;
        if (nt < 15) {
          const _Float16* nxt = W3t + (nt + 1) * 16 * 64;
          a0n = load_a16(nxt, 64, lane, 0);
          a1n = load_a16(nxt, 64, lane, 32);
        }
        v8f acc = {};
        acc = wmma16(a0, b0, acc);
        acc = wmma16(a1, b1v, acc);
        const int fb = nt * 16 + hl * 8;            // 8 contiguous features
        const float4 s0 = *(const float4*)(pS3 + fb);
        const float4 s1v = *(const float4*)(pS3 + fb + 4);
        const float4 o0 = *(const float4*)(pO3 + fb);
        const float4 o1 = *(const float4*)(pO3 + fb + 4);
        v8h hh;
        hh[0] = (_Float16)fmaxf(s0.x * acc[0] + o0.x, 0.f);
        hh[1] = (_Float16)fmaxf(s0.y * acc[1] + o0.y, 0.f);
        hh[2] = (_Float16)fmaxf(s0.z * acc[2] + o0.z, 0.f);
        hh[3] = (_Float16)fmaxf(s0.w * acc[3] + o0.w, 0.f);
        hh[4] = (_Float16)fmaxf(s1v.x * acc[4] + o1.x, 0.f);
        hh[5] = (_Float16)fmaxf(s1v.y * acc[5] + o1.y, 0.f);
        hh[6] = (_Float16)fmaxf(s1v.z * acc[6] + o1.z, 0.f);
        hh[7] = (_Float16)fmaxf(s1v.w * acc[7] + o1.w, 0.f);
        *(v8h*)(sAF + row * 256 + fb) = hh;         // [row][f] row-major (b128 store)
      }
    }

    // ================= Phase 4: logits^T = (W4-tile @ a^T)/sqrt(C) =================
    {
      v16h bReg[8];                                 // hoisted activation B: 64 VGPRs
#pragma unroll
      for (int ks = 0; ks < 8; ++ks) bReg[ks] = load_b16(sAF, 256, lane, ks * 32);
#pragma unroll
      for (int nt = 0; nt < 4; ++nt) {
        const _Float16* base = W4t + nt * 16 * 256;
        v8f acc = {};
#pragma unroll
        for (int ks = 0; ks < 8; ++ks)
          acc = wmma16(load_a16(base, 256, lane, ks * 32), bReg[ks], acc);
        const int cb = nt * 16 + hl * 8;
        const float4 s0 = *(const float4*)(pb4s + cb);
        const float4 s1v = *(const float4*)(pb4s + cb + 4);
        const float4 r0 = make_float4(acc[0] * 0.125f + s0.x,  acc[1] * 0.125f + s0.y,
                                      acc[2] * 0.125f + s0.z,  acc[3] * 0.125f + s0.w);
        const float4 r1 = make_float4(acc[4] * 0.125f + s1v.x, acc[5] * 0.125f + s1v.y,
                                      acc[6] * 0.125f + s1v.z, acc[7] * 0.125f + s1v.w);
        *(float4*)(sQK + row * 64 + cb) = r0;       // logits (f32, b128 stores)
        *(float4*)(sQK + row * 64 + cb + 4) = r1;
      }
    }

    // ================= Phase 5: softmax over channel axis (64) per row =============
    float xv[32];
    float mx = -3.4e38f;
#pragma unroll
    for (int i = 0; i < 32; ++i) { xv[i] = sQK[r * 64 + hf * 32 + i]; mx = fmaxf(mx, xv[i]); }
    mx = fmaxf(mx, __shfl_xor(mx, 1, 32));    // join the two half-row lanes (wave32)
    float sum = 0.f;
#pragma unroll
    for (int i = 0; i < 32; ++i) { xv[i] = __expf(xv[i] - mx); sum += xv[i]; }
    sum += __shfl_xor(sum, 1, 32);
    const float inv = 1.f / sum;
#pragma unroll
    for (int i = 0; i < 32; ++i) sQK[r * 64 + hf * 32 + i] = xv[i] * inv;

    // ================= Phase 6: agg[c] = sum_k a[k][c] * (v+pe)[k][c] ==============
    float a0 = 0.f, a1 = 0.f;
#pragma unroll
    for (int rr = 0; rr < 16; ++rr) {
      a0 += sQK[rr * 64 + lane]      * sVal[rr * 64 + lane];
      a1 += sQK[rr * 64 + lane + 32] * sVal[rr * 64 + lane + 32];
    }
    out[pOff + lane]      = a0;
    out[pOff + lane + 32] = a1;
  }
}

extern "C" void kernel_launch(void* const* d_in, const int* in_sizes, int n_in,
                              void* d_out, int out_size, void* d_ws, size_t ws_size,
                              hipStream_t stream) {
  (void)in_sizes; (void)n_in; (void)out_size; (void)d_ws; (void)ws_size;
  const float* key   = (const float*)d_in[0];
  const float* query = (const float*)d_in[1];
  const float* value = (const float*)d_in[2];
  const float* pos   = (const float*)d_in[3];
  const int*   idx   = (const int*)d_in[4];
  const float* W1  = (const float*)d_in[5];
  const float* b1  = (const float*)d_in[6];
  const float* g1  = (const float*)d_in[7];
  const float* bt1 = (const float*)d_in[8];
  const float* W2  = (const float*)d_in[9];
  const float* b2  = (const float*)d_in[10];
  const float* W3  = (const float*)d_in[11];
  const float* b3  = (const float*)d_in[12];
  const float* g3  = (const float*)d_in[13];
  const float* bt3 = (const float*)d_in[14];
  const float* W4  = (const float*)d_in[15];
  const float* b4  = (const float*)d_in[16];
  float* out = (float*)d_out;

  const int nPoints = BB * NN;                       // 65536 points
  const dim3 grid(nPoints / (WAVES * PPW));          // 1024 WGs x 4 waves x 16 points
  const dim3 block(128);
  const size_t shmem = 77824 + 4 * 20480;            // 159744 B
  pt_attn_wmma<<<grid, block, shmem, stream>>>(key, query, value, pos, idx,
                                               W1, b1, g1, bt1, W2, b2,
                                               W3, b3, g3, bt3, W4, b4, out);
}
